// AttnBlock_5712306504246
// MI455X (gfx1250) — matrix operand order repaired and checked by reading, not compiled, not run on hardware
//
#include <hip/hip_runtime.h>
#include <math.h>

// Problem constants (match reference)
#define B_   16
#define C_   512
#define HW_  1024
#define G_   32
#define CPG_ (C_ / G_)            // 16 channels per group
#define EPS_ 1e-5f

typedef __bf16 bf16;
typedef __attribute__((ext_vector_type(16))) __bf16 v16bf;
typedef __attribute__((ext_vector_type(8)))  __bf16 v8bf;
typedef __attribute__((ext_vector_type(4)))  __bf16 v4bf;
typedef __attribute__((ext_vector_type(8)))  float  v8f;
typedef __attribute__((ext_vector_type(4)))  float  v4f;

// ---------------------------------------------------------------------------
// Kernel 1: convert the four 512x512 fp32 weight matrices to bf16 (row-major,
// (out_ch, in_ch) -> used directly as WMMA "A" operand).
// ---------------------------------------------------------------------------
__global__ void cvt_weights_kernel(const float* __restrict__ w0,
                                   const float* __restrict__ w1,
                                   const float* __restrict__ w2,
                                   const float* __restrict__ w3,
                                   void* o0, void* o1, void* o2, void* o3,
                                   int n) {
  int i = blockIdx.x * blockDim.x + threadIdx.x;
  if (i < n) {
    ((bf16*)o0)[i] = (bf16)w0[i];
    ((bf16*)o1)[i] = (bf16)w1[i];
    ((bf16*)o2)[i] = (bf16)w2[i];
    ((bf16*)o3)[i] = (bf16)w3[i];
  }
}

// ---------------------------------------------------------------------------
// Kernel 2: GroupNorm. One block per (batch, group). fp32 statistics over
// 16 channels x 1024 spatial = 16384 elements (4096 float4), then write the
// normalized result as bf16 TRANSPOSED to hn_t[b][hw][c]  (the N x K layout
// the QKV GEMMs need for their B operand).
// ---------------------------------------------------------------------------
__global__ __launch_bounds__(256)
void groupnorm_kernel(const float* __restrict__ x,
                      const float* __restrict__ scale,
                      const float* __restrict__ bias,
                      void* __restrict__ hn_t_raw) {
  bf16* hn_t = (bf16*)hn_t_raw;
  const int NV4 = CPG_ * HW_ / 4;                     // 4096 float4
  int b = blockIdx.x / G_;
  int g = blockIdx.x % G_;
  const float* xp = x + ((size_t)b * C_ + (size_t)g * CPG_) * HW_;
  const v4f* xp4 = (const v4f*)xp;

  __shared__ float red[256];
  float s = 0.f, ss = 0.f;
  for (int e = threadIdx.x; e < NV4; e += 256) {
    v4f v = xp4[e];
#pragma unroll
    for (int j = 0; j < 4; j++) { s += v[j]; ss += v[j] * v[j]; }
  }
  red[threadIdx.x] = s; __syncthreads();
  for (int o = 128; o > 0; o >>= 1) {
    if (threadIdx.x < o) red[threadIdx.x] += red[threadIdx.x + o];
    __syncthreads();
  }
  float mu = red[0] * (1.f / (CPG_ * HW_));
  __syncthreads();
  red[threadIdx.x] = ss; __syncthreads();
  for (int o = 128; o > 0; o >>= 1) {
    if (threadIdx.x < o) red[threadIdx.x] += red[threadIdx.x + o];
    __syncthreads();
  }
  float var  = red[0] * (1.f / (CPG_ * HW_)) - mu * mu;
  float rinv = rsqrtf(var + EPS_);

  const int HV4 = HW_ / 4;                            // 256 float4 per channel
  for (int e = threadIdx.x; e < NV4; e += 256) {
    int cl  = e / HV4;
    int hw4 = e - cl * HV4;
    int c   = g * CPG_ + cl;
    float sc = scale[c], bi = bias[c];
    v4f v = xp4[e];
    bf16* dst = hn_t + ((size_t)b * HW_ + (size_t)hw4 * 4) * C_ + c;
#pragma unroll
    for (int j = 0; j < 4; j++)
      dst[(size_t)j * C_] = (bf16)((v[j] - mu) * rinv * sc + bi);
  }
}

// ---------------------------------------------------------------------------
// Kernel 3: generic bf16 WMMA GEMM.   D = alpha * A(MxK) * Bt(NxK)^T  [+bias]
// Wave tile 64x64 (4x4 fragments of 16x16), block = 8 waves -> 256x128 tile.
// Fragment loads use one contiguous k order (k = 16h + s) for both A and B:
//   A (16x32 bf16): lane l -> row m=l&15, half h=l>>4; one contiguous v16bf
//     load at K = 16h of a row-major M x K tile.
//   B (32x16 bf16): lane l -> col n=l&15, half h=l>>4; one contiguous v16bf
//     load at K = 16h of a row-major N x K (= B^T) tile.
//   D (16x16 f32):  VGPR r, lane l -> (m = r + 8*(l>>4), n = l&15).
// modes: 0 = f32 D[m][n]              (attention scores, alpha = C^-0.5)
//        1 = bf16 D_t[n][m]           (q_t / k_t / attn-out_t), vector store
//        2 = bf16 D[m][n]             (v)
//        3 = f32 D[m][n] + bias + residual   (final projection + skip)
// ---------------------------------------------------------------------------
__global__ __launch_bounds__(256)
void gemm_bf16_wmma(const void* __restrict__ Araw, long long sA, int lda,
                    const void* __restrict__ Btraw, long long sB, int ldb,
                    void* __restrict__ Draw, long long sD, int ldd,
                    int K, float alpha,
                    const float* __restrict__ bias,
                    const float* __restrict__ resid, long long sR,
                    int mode) {
  const int z = blockIdx.z;
  const bf16* A  = (const bf16*)Araw  + (size_t)z * (size_t)sA;
  const bf16* Bt = (const bf16*)Btraw + (size_t)z * (size_t)sB;

  const int lane = threadIdx.x & 31;
  const int wave = threadIdx.x >> 5;           // 0..7
  const int wm   = wave >> 1;                  // 0..3
  const int wn   = wave & 1;                   // 0..1
  const int m0   = blockIdx.y * 256 + wm * 64;
  const int n0   = blockIdx.x * 128 + wn * 64;

  const int mrow = lane & 15;
  const int h    = lane >> 4;
  const int ncol = lane & 15;

  v8f zero = {};
  v8f acc[4][4];
#pragma unroll
  for (int i = 0; i < 4; i++)
#pragma unroll
    for (int j = 0; j < 4; j++) acc[i][j] = zero;

  for (int k0 = 0; k0 < K; k0 += 32) {
    v16bf af[4], bfr[4];
#pragma unroll
    for (int mi = 0; mi < 4; mi++) {
      const bf16* pa = A + (size_t)(m0 + mi * 16 + mrow) * lda + k0 + 16 * h;
      af[mi] = *(const v16bf*)pa;         // K = 16h .. 16h+15
    }
#pragma unroll
    for (int ni = 0; ni < 4; ni++) {
      const bf16* pb = Bt + (size_t)(n0 + ni * 16 + ncol) * ldb + k0 + 16 * h;
      bfr[ni] = *(const v16bf*)pb;        // K = 16h .. 16h+15
    }
#pragma unroll
    for (int mi = 0; mi < 4; mi++)
#pragma unroll
      for (int ni = 0; ni < 4; ni++)
        acc[mi][ni] = __builtin_amdgcn_wmma_f32_16x16x32_bf16(
            false, af[mi], false, bfr[ni], (short)0, acc[mi][ni],
            false, false);
  }

  float* Df = (float*)Draw + (size_t)z * (size_t)sD;
  bf16*  Db = (bf16*)Draw  + (size_t)z * (size_t)sD;
  const float* R = resid ? (resid + (size_t)z * (size_t)sR) : nullptr;

  if (mode == 1) {
    // Transposed bf16 store: per fragment, one lane owns 8 consecutive m at
    // fixed n -> single 16B store.
#pragma unroll
    for (int mi = 0; mi < 4; mi++)
#pragma unroll
      for (int ni = 0; ni < 4; ni++) {
        int mb = m0 + mi * 16 + 8 * h;
        int n  = n0 + ni * 16 + ncol;
        v8bf pk;
#pragma unroll
        for (int r = 0; r < 8; r++) {
          float v = alpha * acc[mi][ni][r];
          if (bias) v += bias[mb + r];
          pk[r] = (bf16)v;
        }
        *(v8bf*)(Db + (size_t)n * ldd + mb) = pk;
      }
  } else {
#pragma unroll
    for (int mi = 0; mi < 4; mi++)
#pragma unroll
      for (int ni = 0; ni < 4; ni++)
#pragma unroll
        for (int r = 0; r < 8; r++) {
          int m = m0 + mi * 16 + r + 8 * h;
          int n = n0 + ni * 16 + ncol;
          float v = alpha * acc[mi][ni][r];
          if (bias) v += bias[m];
          if (mode == 0)      Df[(size_t)m * ldd + n] = v;
          else if (mode == 2) Db[(size_t)m * ldd + n] = (bf16)v;
          else                Df[(size_t)m * ldd + n] = v + R[(size_t)m * ldd + n];
        }
  }
}

// ---------------------------------------------------------------------------
// Kernel 4: row softmax, in place S(f32, 1024) -> P(bf16) written over the
// first half of the same row (row byte stride stays 4096; P row = 2048B).
// One block per row, one float4 per thread. All reads complete before the
// first barrier, all writes happen after the last barrier -> no hazard.
// ---------------------------------------------------------------------------
__global__ __launch_bounds__(256)
void softmax_rows_kernel(float* __restrict__ S) {
  size_t row = blockIdx.x;
  float* sr = S + row * HW_;
  v4f vals = ((const v4f*)sr)[threadIdx.x];
  float mx = fmaxf(fmaxf(vals[0], vals[1]), fmaxf(vals[2], vals[3]));

  __shared__ float red[256];
  red[threadIdx.x] = mx; __syncthreads();
  for (int o = 128; o > 0; o >>= 1) {
    if (threadIdx.x < o) red[threadIdx.x] = fmaxf(red[threadIdx.x], red[threadIdx.x + o]);
    __syncthreads();
  }
  mx = red[0]; __syncthreads();

  float sum = 0.f;
#pragma unroll
  for (int i = 0; i < 4; i++) {
    vals[i] = __expf(vals[i] - mx);
    sum += vals[i];
  }
  red[threadIdx.x] = sum; __syncthreads();
  for (int o = 128; o > 0; o >>= 1) {
    if (threadIdx.x < o) red[threadIdx.x] += red[threadIdx.x + o];
    __syncthreads();
  }
  float rs = 1.f / red[0];

  v4bf pk;
#pragma unroll
  for (int i = 0; i < 4; i++) pk[i] = (bf16)(vals[i] * rs);
  ((v4bf*)sr)[threadIdx.x] = pk;     // 8B packed store
}

// ---------------------------------------------------------------------------
// Host launcher
// ---------------------------------------------------------------------------
extern "C" void kernel_launch(void* const* d_in, const int* in_sizes, int n_in,
                              void* d_out, int out_size, void* d_ws, size_t ws_size,
                              hipStream_t stream) {
  const float* x   = (const float*)d_in[0];
  // d_in[1] = temb: unused by the reference
  const float* gns = (const float*)d_in[2];
  const float* gnb = (const float*)d_in[3];
  const float* wq  = (const float*)d_in[4];
  const float* bq  = (const float*)d_in[5];
  const float* wk  = (const float*)d_in[6];
  const float* bk  = (const float*)d_in[7];
  const float* wv  = (const float*)d_in[8];
  const float* bv  = (const float*)d_in[9];
  const float* wo  = (const float*)d_in[10];
  const float* bo  = (const float*)d_in[11];

  // Workspace layout (bytes). Total ~146 MB.
  char* ws = (char*)d_ws;
  const size_t WSZ   = (size_t)C_ * C_ * 2;                 // one bf16 weight: 512 KB
  const size_t TSZ   = (size_t)B_ * HW_ * C_ * 2;           // bf16 (B,HW,C): 16 MB
  char* wq_b = ws;                    // 4 weights
  char* wk_b = wq_b + WSZ;
  char* wv_b = wk_b + WSZ;
  char* wo_b = wv_b + WSZ;
  char* hn_t = wo_b + WSZ;            // (B, HW, C) bf16
  char* q_t  = hn_t + TSZ;            // (B, HW, C) bf16
  char* k_t  = q_t  + TSZ;            // (B, HW, C) bf16
  char* v_m  = k_t  + TSZ;            // (B, C, HW) bf16
  char* o2_t = v_m  + TSZ;            // (B, HW, C) bf16
  char* S    = o2_t + TSZ;            // (B, HW, HW) f32; P bf16 in-place

  // 1) weights -> bf16
  {
    int n = C_ * C_;
    cvt_weights_kernel<<<(n + 255) / 256, 256, 0, stream>>>(
        wq, wk, wv, wo, wq_b, wk_b, wv_b, wo_b, n);
  }

  // 2) GroupNorm -> hn_t (B,HW,C) bf16
  groupnorm_kernel<<<B_ * G_, 256, 0, stream>>>(x, gns, gnb, hn_t);

  dim3 blk(256);
  const long long sHN = (long long)HW_ * C_;

  // 3) Q = Wq * HN  -> q_t[hw][c]   (M=C, N=HW, K=C)
  {
    dim3 grid(HW_ / 128, C_ / 256, B_);
    gemm_bf16_wmma<<<grid, blk, 0, stream>>>(
        wq_b, 0, C_, hn_t, sHN, C_, q_t, sHN, C_, C_, 1.f, bq, nullptr, 0, 1);
    gemm_bf16_wmma<<<grid, blk, 0, stream>>>(
        wk_b, 0, C_, hn_t, sHN, C_, k_t, sHN, C_, C_, 1.f, bk, nullptr, 0, 1);
    // V kept channel-major: v[c][hw]
    gemm_bf16_wmma<<<grid, blk, 0, stream>>>(
        wv_b, 0, C_, hn_t, sHN, C_, v_m, (long long)C_ * HW_, HW_,
        C_, 1.f, bv, nullptr, 0, 2);
  }

  // 4) S = (Q^T K) * C^-0.5   (M=HW, N=HW, K=C), f32
  {
    dim3 grid(HW_ / 128, HW_ / 256, B_);
    const float scale = 0.044194173824159216f;   // 512^-0.5
    gemm_bf16_wmma<<<grid, blk, 0, stream>>>(
        q_t, sHN, C_, k_t, sHN, C_, S, (long long)HW_ * HW_, HW_,
        C_, scale, nullptr, nullptr, 0, 0);
  }

  // 5) softmax rows, S(f32) -> P(bf16) in place
  softmax_rows_kernel<<<B_ * HW_, 256, 0, stream>>>((float*)S);

  // 6) O2 = V * P^T  -> o2_t[i][c]   (M=C, N=HW(i), K=HW(j))
  //    P rows live at byte stride 4096 -> ldb = 2*HW_ bf16 elements,
  //    batch stride = 2*HW_*HW_ bf16 elements.
  {
    dim3 grid(HW_ / 128, C_ / 256, B_);
    gemm_bf16_wmma<<<grid, blk, 0, stream>>>(
        v_m, (long long)C_ * HW_, HW_,
        S, (long long)2 * HW_ * HW_, 2 * HW_,
        o2_t, sHN, C_,
        HW_, 1.f, nullptr, nullptr, 0, 1);
  }

  // 7) out = Wo * O2 + bo + x   (M=C, N=HW, K=C), f32 to d_out
  {
    dim3 grid(HW_ / 128, C_ / 256, B_);
    gemm_bf16_wmma<<<grid, blk, 0, stream>>>(
        wo_b, 0, C_, o2_t, sHN, C_,
        d_out, (long long)C_ * HW_, HW_,
        C_, 1.f, bo, x, (long long)C_ * HW_, 3);
  }
}
